// RoPEMultiheadAttention_48498770707176
// MI455X (gfx1250) — matrix operand order repaired and checked by reading, not compiled, not run on hardware
//
#include <hip/hip_runtime.h>

// ---------------------------------------------------------------------------
// CDNA5 (gfx1250) wave32 WMMA types & helpers
// ---------------------------------------------------------------------------
typedef __attribute__((ext_vector_type(16))) __bf16 v16bf;
typedef __attribute__((ext_vector_type(8)))  float  v8f;

union Frag { v16bf v; unsigned int u[8]; };

__device__ __forceinline__ unsigned short f2bf(float f) {
    unsigned int x = __builtin_bit_cast(unsigned int, f);
    x += 0x7fffu + ((x >> 16) & 1u);          // round-to-nearest-even
    return (unsigned short)(x >> 16);
}

// A-matrix (16x32, bf16) k-offset for VGPR j, lane-half hl (ISA 7.12.2)
__device__ __forceinline__ int kA(int j, int hl) {
    return ((j >> 2) << 4) | (hl << 3) | ((j & 3) << 1);
}
// B-matrix (32x16, bf16) k-offset for VGPR j, lane-half hl (same k order as A)
__device__ __forceinline__ int kB(int j, int hl) {
    return ((j >> 2) << 4) | (hl << 3) | ((j & 3) << 1);
}

__device__ __forceinline__ v8f wmma_bf16(v16bf a, v16bf b, v8f c) {
    return __builtin_amdgcn_wmma_f32_16x16x32_bf16(
        /*neg_a=*/false, a, /*neg_b=*/false, b,
        /*c_mod=*/(short)0, c, /*reuse_a=*/false, /*reuse_b=*/false);
}

// ---------------------------------------------------------------------------
// Elementwise pack / convert kernels
// ---------------------------------------------------------------------------
__global__ void cvt_bf16(const float* __restrict__ s, unsigned short* __restrict__ d, int count) {
    int i = blockIdx.x * blockDim.x + threadIdx.x;
    if (i < count) d[i] = f2bf(s[i]);
}

// global_feat (n, 512, 4096) -> gf_bf (n, 4096, 512) bf16
__global__ void pack_gf(const float* __restrict__ g, unsigned short* __restrict__ dst, int total) {
    int i = blockIdx.x * blockDim.x + threadIdx.x;
    if (i >= total) return;
    int c = i & 511;
    int p = (i >> 9) & 4095;
    int b = i >> 21;
    dst[i] = f2bf(g[((size_t)(b * 512 + c)) * 4096 + p]);
}

// V f32 (n, 4096, 512) -> Vt bf16 (n, 8, 64, 4096)  (per-head dh-major)
__global__ void pack_vt(const float* __restrict__ v, unsigned short* __restrict__ dst, int total) {
    int i = blockIdx.x * blockDim.x + threadIdx.x;
    if (i >= total) return;
    int p  = i & 4095;
    int c  = (i >> 12) & 63;
    int hh = (i >> 18) & 7;
    int b  = i >> 21;
    dst[i] = f2bf(v[((size_t)(b * 4096 + p)) * 512 + hh * 64 + c]);
}

// RoPE + f32 -> bf16.  src layout (n, ntok, 512) viewed as (n, ntok, 8, 64).
// thread handles dh pair (i, i+32) for one (b, tok, head).
__global__ void rope_pack(const float* __restrict__ src, unsigned short* __restrict__ dst,
                          int ntok, int gridw, float invg, int total) {
    int tid = blockIdx.x * blockDim.x + threadIdx.x;
    if (tid >= total) return;
    int i    = tid & 31;
    int head = (tid >> 5) & 7;
    int tq   = tid >> 8;
    int tok  = tq % ntok;
    int b    = tq / ntok;
    // freq[i] = 10000^(-i/32) = exp(-i * ln(10000)/32)
    float freq = __expf(-0.28782313662425572f * (float)i);
    float x = (float)(tok % gridw) * invg;
    float y = (float)(tok / gridw) * invg;
    float sx, cx, sy, cy;
    __sincosf(x * freq, &sx, &cx);
    __sincosf(y * freq, &sy, &cy);
    size_t base = ((size_t)(b * ntok + tok)) * 512 + head * 64 + i;
    float t0 = src[base], t1 = src[base + 32];
    dst[base]      = f2bf(t0 * cx - t1 * sx);   // out[i]    = t[i]c - t[i+32]s
    dst[base + 32] = f2bf(t1 * cy + t0 * sy);   // out[i+32] = t[i+32]c + t[i]s
}

// enhanced = local + sigmoid(gate_logits) * ao  -> bf16
__global__ void gate_fin(const float* __restrict__ gl, const float* __restrict__ loc,
                         const float* __restrict__ ao, unsigned short* __restrict__ enh, int count) {
    int i = blockIdx.x * blockDim.x + threadIdx.x;
    if (i >= count) return;
    float g = 1.0f / (1.0f + __expf(-gl[i]));
    enh[i] = f2bf(loc[i] + g * ao[i]);
}

// ---------------------------------------------------------------------------
// Generic WMMA GEMM:  C[M,N](f32) = A[M,K](bf16) @ W[N,K]^T(bf16) (+bias)(+C)
// 256 threads = 8 waves; each wave computes a 16x64 tile; block covers 128x64.
// grid.x = (M/128) * (N/64)
// ---------------------------------------------------------------------------
__global__ __launch_bounds__(256) void gemm_xwT(
    const unsigned short* __restrict__ A, int lda,
    const unsigned short* __restrict__ W, int ldw,
    float* __restrict__ C, int ldc,
    const float* __restrict__ bias, int N, int K, int accum) {
    const int lane = threadIdx.x & 31;
    const int wave = threadIdx.x >> 5;
    const int r  = lane & 15;
    const int hl = lane >> 4;
    const int tilesN = N >> 6;
    const int m0 = (blockIdx.x / tilesN) * 128 + wave * 16;
    const int n0 = (blockIdx.x % tilesN) * 64;

    v8f acc[4] = {};
    const unsigned short* Arow = A + (size_t)(m0 + r) * lda;

    for (int kk = 0; kk < K; kk += 32) {
        Frag a, bfr[4];
#pragma unroll
        for (int j = 0; j < 8; ++j)
            a.u[j] = *(const unsigned int*)&Arow[kk + kA(j, hl)];
#pragma unroll
        for (int t = 0; t < 4; ++t) {
            const unsigned short* Wrow = W + (size_t)(n0 + t * 16 + r) * ldw + kk;
#pragma unroll
            for (int j = 0; j < 8; ++j)
                bfr[t].u[j] = *(const unsigned int*)&Wrow[kB(j, hl)];
        }
#pragma unroll
        for (int t = 0; t < 4; ++t)
            acc[t] = wmma_bf16(a.v, bfr[t].v, acc[t]);
    }

#pragma unroll
    for (int t = 0; t < 4; ++t) {
        const int col = n0 + t * 16 + r;
        const float bv = bias ? bias[col] : 0.0f;
#pragma unroll
        for (int vv = 0; vv < 8; ++vv) {
            const size_t idx = (size_t)(m0 + vv + (hl << 3)) * ldc + col;
            float val = acc[t][vv] + bv;
            if (accum) val += C[idx];
            C[idx] = val;
        }
    }
}

// ---------------------------------------------------------------------------
// Flash attention: one wave per (b, h, 16-query tile). 4096 keys, 32/iter.
// Qr/Kr: rope'd bf16 in (n, tok, 512); Vt: bf16 (n, 8, 64, 4096).
// ---------------------------------------------------------------------------
__global__ __launch_bounds__(32) void attn_flash(
    const unsigned short* __restrict__ Qr,
    const unsigned short* __restrict__ Kr,
    const unsigned short* __restrict__ Vt,
    float* __restrict__ aof, unsigned short* __restrict__ aob) {
    __shared__ unsigned short pst[16 * 32];   // P-tile staging (transpose via LDS)

    const int lane = threadIdx.x & 31;
    const int r  = lane & 15;
    const int hl = lane >> 4;
    const int q0 = (blockIdx.x & 63) << 4;    // Np/16 = 64 query tiles
    const int h  = (blockIdx.x >> 6) & 7;
    const int b  = blockIdx.x >> 9;

    // Q fragments (A operand): 16 rows x dh=64 -> two 16x32 steps
    Frag qf[2];
    {
        const size_t qb = ((size_t)(b * 1024 + q0 + r)) * 512 + h * 64;
#pragma unroll
        for (int s = 0; s < 2; ++s)
#pragma unroll
            for (int j = 0; j < 8; ++j)
                qf[s].u[j] = *(const unsigned int*)&Qr[qb + s * 32 + kA(j, hl)];
    }

    float rowm[8], rowl[8];
#pragma unroll
    for (int vv = 0; vv < 8; ++vv) { rowm[vv] = -1e30f; rowl[vv] = 0.0f; }
    v8f acc[4] = {};

    const size_t kbad = ((size_t)b * 4096) * 512 + h * 64;          // + key*512
    const size_t vbad = ((size_t)(b * 8 + h)) * 64 * 4096;          // + c*4096 + key

    for (int kt = 0; kt < 4096; kt += 32) {
        // K^T fragments (B operand): key = column; dh contiguous in memory.
        Frag kf0[2], kf1[2];
#pragma unroll
        for (int s = 0; s < 2; ++s) {
            const size_t k0 = kbad + (size_t)(kt + r) * 512 + s * 32;
            const size_t k1 = kbad + (size_t)(kt + 16 + r) * 512 + s * 32;
#pragma unroll
            for (int j = 0; j < 8; ++j) {
                kf0[s].u[j] = *(const unsigned int*)&Kr[k0 + kB(j, hl)];
                kf1[s].u[j] = *(const unsigned int*)&Kr[k1 + kB(j, hl)];
            }
        }
        v8f s0 = {}, s1 = {};
        s0 = wmma_bf16(qf[0].v, kf0[0].v, s0);
        s0 = wmma_bf16(qf[1].v, kf0[1].v, s0);
        s1 = wmma_bf16(qf[0].v, kf1[0].v, s1);
        s1 = wmma_bf16(qf[1].v, kf1[1].v, s1);

        // online softmax: row stats reduced across 16-lane halves (wave32)
#pragma unroll
        for (int vv = 0; vv < 8; ++vv) {
            float e0 = s0[vv] * 0.125f;       // 1/sqrt(64)
            float e1 = s1[vv] * 0.125f;
            float tm = fmaxf(e0, e1);
            tm = fmaxf(tm, __shfl_xor(tm, 1, 32));
            tm = fmaxf(tm, __shfl_xor(tm, 2, 32));
            tm = fmaxf(tm, __shfl_xor(tm, 4, 32));
            tm = fmaxf(tm, __shfl_xor(tm, 8, 32));
            float mnew  = fmaxf(rowm[vv], tm);
            float alpha = __expf(rowm[vv] - mnew);
            rowm[vv] = mnew;
            float p0 = __expf(e0 - mnew);
            float p1 = __expf(e1 - mnew);
            float ts = p0 + p1;
            ts += __shfl_xor(ts, 1, 32);
            ts += __shfl_xor(ts, 2, 32);
            ts += __shfl_xor(ts, 4, 32);
            ts += __shfl_xor(ts, 8, 32);
            rowl[vv] = rowl[vv] * alpha + ts;
#pragma unroll
            for (int t = 0; t < 4; ++t) acc[t][vv] *= alpha;
            const int m = vv + (hl << 3);     // C-layout row for this element
            pst[m * 32 + r]      = f2bf(p0);
            pst[m * 32 + 16 + r] = f2bf(p1);
        }
        // CDNA5 split counter: make the LDS stores visible before A-frag reads
        asm volatile("s_wait_dscnt 0x0" ::: "memory");

        Frag pf;                              // P as A operand (16x32 bf16)
#pragma unroll
        for (int j = 0; j < 8; ++j)
            pf.u[j] = *(const unsigned int*)&pst[r * 32 + kA(j, hl)];

        Frag vf[4];                           // V^T as B operand (keys contiguous)
#pragma unroll
        for (int t = 0; t < 4; ++t) {
            const size_t vb = vbad + (size_t)(t * 16 + r) * 4096 + kt;
#pragma unroll
            for (int j = 0; j < 8; ++j)
                vf[t].u[j] = *(const unsigned int*)&Vt[vb + kB(j, hl)];
        }
#pragma unroll
        for (int t = 0; t < 4; ++t)
            acc[t] = wmma_bf16(pf.v, vf[t].v, acc[t]);
    }

#pragma unroll
    for (int vv = 0; vv < 8; ++vv) {
        const float inv = 1.0f / rowl[vv];
        const int m = vv + (hl << 3);
        const size_t ob = ((size_t)(b * 1024 + q0 + m)) * 512 + h * 64;
#pragma unroll
        for (int t = 0; t < 4; ++t) {
            float val = acc[t][vv] * inv;
            aof[ob + t * 16 + r] = val;
            aob[ob + t * 16 + r] = f2bf(val);
        }
    }
}

// ---------------------------------------------------------------------------
// Host orchestration
// ---------------------------------------------------------------------------
extern "C" void kernel_launch(void* const* d_in, const int* in_sizes, int n_in,
                              void* d_out, int out_size, void* d_ws, size_t ws_size,
                              hipStream_t stream) {
    (void)in_sizes; (void)n_in; (void)out_size; (void)ws_size;
    const float* local = (const float*)d_in[0];
    const float* gfeat = (const float*)d_in[1];
    const float* Wq = (const float*)d_in[2];
    const float* bq = (const float*)d_in[3];
    const float* Wk = (const float*)d_in[4];
    const float* bk = (const float*)d_in[5];
    const float* Wv = (const float*)d_in[6];
    const float* bv = (const float*)d_in[7];
    const float* Wg = (const float*)d_in[8];
    const float* bg = (const float*)d_in[9];
    const float* Wo = (const float*)d_in[10];
    const float* bo = (const float*)d_in[11];

    char* w = (char*)d_ws;
    size_t off = 0;
    auto take = [&](size_t bytes) {
        void* p = w + off;
        off += (bytes + 255) & ~(size_t)255;
        return p;
    };
    const size_t ML = 4ull * 1024;   // local rows
    const size_t MG = 4ull * 4096;   // global rows

    unsigned short* lf_bf = (unsigned short*)take(ML * 512 * 2);
    unsigned short* gf_bf = (unsigned short*)take(MG * 512 * 2);
    unsigned short* wq_bf = (unsigned short*)take(512ull * 512 * 2);
    unsigned short* wk_bf = (unsigned short*)take(512ull * 512 * 2);
    unsigned short* wv_bf = (unsigned short*)take(512ull * 512 * 2);
    unsigned short* wo_bf = (unsigned short*)take(512ull * 512 * 2);
    unsigned short* wg_bf = (unsigned short*)take(512ull * 1024 * 2);
    float* qf32 = (float*)take(ML * 512 * 4);          // later reused: gate logits
    float* kf32 = (float*)take(MG * 512 * 4);          // later reused: ao f32
    float* vf32 = (float*)take(MG * 512 * 4);          // later reused: ao bf16 + enh bf16
    unsigned short* qr = (unsigned short*)take(ML * 512 * 2);
    unsigned short* kr = (unsigned short*)take(MG * 512 * 2);
    unsigned short* vt = (unsigned short*)take(MG * 512 * 2);

    float* gate_f32        = qf32;                                   // Q consumed by rope
    float* ao_f32          = kf32;                                   // K consumed by rope
    unsigned short* ao_bf  = (unsigned short*)vf32;                  // V consumed by pack_vt
    unsigned short* enh_bf = (unsigned short*)((char*)vf32 + ML * 512 * 2);

    const int T = 256;
    // pack inputs / weights to bf16
    cvt_bf16<<<(int)((ML * 512 + T - 1) / T), T, 0, stream>>>(local, lf_bf, (int)(ML * 512));
    cvt_bf16<<<(512 * 512 + T - 1) / T, T, 0, stream>>>(Wq, wq_bf, 512 * 512);
    cvt_bf16<<<(512 * 512 + T - 1) / T, T, 0, stream>>>(Wk, wk_bf, 512 * 512);
    cvt_bf16<<<(512 * 512 + T - 1) / T, T, 0, stream>>>(Wv, wv_bf, 512 * 512);
    cvt_bf16<<<(512 * 512 + T - 1) / T, T, 0, stream>>>(Wo, wo_bf, 512 * 512);
    cvt_bf16<<<(512 * 1024 + T - 1) / T, T, 0, stream>>>(Wg, wg_bf, 512 * 1024);
    pack_gf<<<(int)((MG * 512 + T - 1) / T), T, 0, stream>>>(gfeat, gf_bf, (int)(MG * 512));

    // projections (WMMA GEMMs): grid = (M/128)*(N/64)
    gemm_xwT<<<(int)(ML / 128) * 8, 256, 0, stream>>>(lf_bf, 512, wq_bf, 512, qf32, 512, bq, 512, 512, 0);
    gemm_xwT<<<(int)(MG / 128) * 8, 256, 0, stream>>>(gf_bf, 512, wk_bf, 512, kf32, 512, bk, 512, 512, 0);
    gemm_xwT<<<(int)(MG / 128) * 8, 256, 0, stream>>>(gf_bf, 512, wv_bf, 512, vf32, 512, bv, 512, 512, 0);

    // RoPE + bf16 pack (local grid 32x32 -> inv 1/31; global 64x64 -> 1/63)
    rope_pack<<<(int)((ML * 256 + T - 1) / T), T, 0, stream>>>(qf32, qr, 1024, 32, 1.0f / 31.0f, (int)(ML * 256));
    rope_pack<<<(int)((MG * 256 + T - 1) / T), T, 0, stream>>>(kf32, kr, 4096, 64, 1.0f / 63.0f, (int)(MG * 256));
    pack_vt<<<(int)((MG * 512 + T - 1) / T), T, 0, stream>>>(vf32, vt, (int)(MG * 512));

    // flash attention: n*h*(Np/16) = 4*8*64 = 2048 waves
    attn_flash<<<2048, 32, 0, stream>>>(qr, kr, vt, ao_f32, ao_bf);

    // gate = sigmoid(local@Wg1^T + ao@Wg2^T + bg); enhanced = local + gate*ao
    gemm_xwT<<<(int)(ML / 128) * 8, 256, 0, stream>>>(lf_bf, 512, wg_bf, 1024, gate_f32, 512, bg, 512, 512, 0);
    gemm_xwT<<<(int)(ML / 128) * 8, 256, 0, stream>>>(ao_bf, 512, wg_bf + 512, 1024, gate_f32, 512, nullptr, 512, 512, 1);
    gate_fin<<<(int)((ML * 512 + T - 1) / T), T, 0, stream>>>(gate_f32, local, ao_f32, enh_bf, (int)(ML * 512));

    // output projection -> d_out (f32)
    gemm_xwT<<<(int)(ML / 128) * 8, 256, 0, stream>>>(enh_bf, 512, wo_bf, 512, (float*)d_out, 512, bo, 512, 512, 0);
}
